// MultiHeadAttention_49426483642562
// MI455X (gfx1250) — matrix operand order repaired and checked by reading, not compiled, not run on hardware
//
#include <hip/hip_runtime.h>

typedef __attribute__((ext_vector_type(16))) __bf16 v16bf;
typedef __attribute__((ext_vector_type(8)))  float  v8f;

#define B_   2
#define L_   4096
#define D_   768
#define H_   12
#define HS_  64
#define SCALE_ 0.125f          // 64^-0.5 (exact power of two)
#define FA_WAVES 4

union FragBF {
    uint4 u[2];
    v16bf v;
};

// ---------------------------------------------------------------------------
// 16-lane row butterfly reductions via DPP16 (no LDS / ds_bpermute):
// xor1 (quad_perm 0xB1), xor2 (quad_perm 0x4E), ^7 (row_half_mirror 0x141),
// ^15 (row_mirror 0x140): pair -> quad -> oct -> full 16-lane row.
// ---------------------------------------------------------------------------
__device__ __forceinline__ float row16_max(float x) {
    x = fmaxf(x, __int_as_float(__builtin_amdgcn_update_dpp(0, __float_as_int(x), 0xB1,  0xF, 0xF, true)));
    x = fmaxf(x, __int_as_float(__builtin_amdgcn_update_dpp(0, __float_as_int(x), 0x4E,  0xF, 0xF, true)));
    x = fmaxf(x, __int_as_float(__builtin_amdgcn_update_dpp(0, __float_as_int(x), 0x141, 0xF, 0xF, true)));
    x = fmaxf(x, __int_as_float(__builtin_amdgcn_update_dpp(0, __float_as_int(x), 0x140, 0xF, 0xF, true)));
    return x;
}
__device__ __forceinline__ float row16_sum(float x) {
    x += __int_as_float(__builtin_amdgcn_update_dpp(0, __float_as_int(x), 0xB1,  0xF, 0xF, true));
    x += __int_as_float(__builtin_amdgcn_update_dpp(0, __float_as_int(x), 0x4E,  0xF, 0xF, true));
    x += __int_as_float(__builtin_amdgcn_update_dpp(0, __float_as_int(x), 0x141, 0xF, 0xF, true));
    x += __int_as_float(__builtin_amdgcn_update_dpp(0, __float_as_int(x), 0x140, 0xF, 0xF, true));
    return x;
}

// ---------------------------------------------------------------------------
// Kernel 0: one-shot f32 -> bf16 conversion (8 elements / thread)
// ---------------------------------------------------------------------------
__global__ __launch_bounds__(256) void cvt_bf16_kernel(const float* __restrict__ in,
                                                       __bf16* __restrict__ out,
                                                       int n8) {
    const int i = blockIdx.x * 256 + threadIdx.x;
    if (i >= n8) return;
    const float4* p = (const float4*)in + (size_t)i * 2;
    const float4 a = p[0], b = p[1];
    union { uint4 u; __bf16 h[8]; } r;
    r.h[0] = (__bf16)a.x; r.h[1] = (__bf16)a.y; r.h[2] = (__bf16)a.z; r.h[3] = (__bf16)a.w;
    r.h[4] = (__bf16)b.x; r.h[5] = (__bf16)b.y; r.h[6] = (__bf16)b.z; r.h[7] = (__bf16)b.w;
    ((uint4*)out)[i] = r.u;
}

// ---------------------------------------------------------------------------
// GEMM: C[16x64 tile] = A_bf16 @ W_bf16^T, one wave per tile, 4 WMMA / k-step.
// QSCATTER=true  : write bf16 into q layout [B,H,L,HS]
// QSCATTER=false : write f32 row-major [8192, 768]
// ---------------------------------------------------------------------------
template<bool QSCATTER>
__global__ __launch_bounds__(256) void gemm16x64_kernel(const __bf16* __restrict__ Ah,
                                                        const __bf16* __restrict__ Wh,
                                                        __bf16* __restrict__ OutH,
                                                        float* __restrict__ OutF) {
    const int wid  = blockIdx.x * 8 + (threadIdx.x >> 5);
    const int lane = threadIdx.x & 31;
    const int half = lane >> 4, l16 = lane & 15;
    const int nt = wid % (D_ / 64);
    const int mt = wid / (D_ / 64);
    const int m0 = mt * 16, n0 = nt * 64;

    const uint4* ap  = (const uint4*)(Ah + (size_t)(m0 + l16) * D_);
    const uint4* bp0 = (const uint4*)(Wh + (size_t)(n0 +  0 + l16) * D_);
    const uint4* bp1 = (const uint4*)(Wh + (size_t)(n0 + 16 + l16) * D_);
    const uint4* bp2 = (const uint4*)(Wh + (size_t)(n0 + 32 + l16) * D_);
    const uint4* bp3 = (const uint4*)(Wh + (size_t)(n0 + 48 + l16) * D_);

    v8f acc0 = {}, acc1 = {}, acc2 = {}, acc3 = {};
    for (int k8 = 0; k8 < D_ / 8; k8 += 4) {          // k0 = 8*k8, step 32
        FragBF a, b0, b1, b2, b3;
        a.u[0]  = ap[k8 + half * 2];    a.u[1]  = ap[k8 + half * 2 + 1];
        b0.u[0] = bp0[k8 + half * 2];   b0.u[1] = bp0[k8 + half * 2 + 1];
        b1.u[0] = bp1[k8 + half * 2];   b1.u[1] = bp1[k8 + half * 2 + 1];
        b2.u[0] = bp2[k8 + half * 2];   b2.u[1] = bp2[k8 + half * 2 + 1];
        b3.u[0] = bp3[k8 + half * 2];   b3.u[1] = bp3[k8 + half * 2 + 1];
        acc0 = __builtin_amdgcn_wmma_f32_16x16x32_bf16(false, a.v, false, b0.v, (short)0, acc0, false, false);
        acc1 = __builtin_amdgcn_wmma_f32_16x16x32_bf16(false, a.v, false, b1.v, (short)0, acc1, false, false);
        acc2 = __builtin_amdgcn_wmma_f32_16x16x32_bf16(false, a.v, false, b2.v, (short)0, acc2, false, false);
        acc3 = __builtin_amdgcn_wmma_f32_16x16x32_bf16(false, a.v, false, b3.v, (short)0, acc3, false, false);
    }

#define GEMM_EPI(T, ACC)                                                          \
    _Pragma("unroll")                                                             \
    for (int i = 0; i < 8; ++i) {                                                 \
        const int row = m0 + i + half * 8;                                        \
        const int col = n0 + (T) * 16 + l16;                                      \
        if (QSCATTER) {                                                           \
            const int bb = row >> 12, l = row & (L_ - 1);                         \
            const int hh = col >> 6,  hs = col & (HS_ - 1);                       \
            OutH[(((size_t)bb * H_ + hh) * L_ + l) * HS_ + hs] = (__bf16)ACC[i];  \
        } else {                                                                  \
            OutF[(size_t)row * D_ + col] = ACC[i];                                \
        }                                                                         \
    }
    GEMM_EPI(0, acc0)
    GEMM_EPI(1, acc1)
    GEMM_EPI(2, acc2)
    GEMM_EPI(3, acc3)
#undef GEMM_EPI
}

// ---------------------------------------------------------------------------
// Flash attention, K = V = Q (reference bug preserved).
// One wave per (b, h, 16-query tile); streams 32-key tiles with online softmax.
// V tiles double-buffered in LDS via GLOBAL_LOAD_ASYNC_TO_LDS_B128 so the
// fetch of tile kt+32 overlaps S/softmax/PV of tile kt. Output bf16 [B,L,D].
// ---------------------------------------------------------------------------
__global__ __launch_bounds__(32 * FA_WAVES) void flash_kernel(
        const __bf16* __restrict__ Qh, __bf16* __restrict__ O) {
    __shared__ __align__(16) __bf16 sP[FA_WAVES][16][32];
    __shared__ __align__(16) __bf16 sV[FA_WAVES][2][32][64];   // double buffer

    const int w    = threadIdx.x >> 5;
    const int lane = threadIdx.x & 31;
    const int half = lane >> 4, l16 = lane & 15;

    int wid = blockIdx.x * FA_WAVES + w;
    const int qt = wid % (L_ / 16); wid /= (L_ / 16);
    const int h  = wid % H_;
    const int b  = wid / H_;
    const __bf16* qp = Qh + ((size_t)(b * H_ + h) * L_) * HS_;
    const int qbase = qt * 16;

    const uint32_t vbase = (uint32_t)(uintptr_t)&sV[w][0][0][0];
    const uint32_t vlane = (uint32_t)lane * (HS_ * 2);         // lane's row offset

#define ISSUE_VTILE(SLOT, ROW0)                                                   \
    {                                                                             \
        const uint64_t ga_ = (uint64_t)(uintptr_t)(qp + (size_t)((ROW0) + lane) * HS_); \
        const uint32_t la_ = vbase + (uint32_t)(SLOT) * (32 * HS_ * 2) + vlane;   \
        _Pragma("unroll")                                                         \
        for (int j_ = 0; j_ < 8; ++j_) {                                          \
            asm volatile("global_load_async_to_lds_b128 %0, %1, off"              \
                         :: "v"(la_ + j_ * 16), "v"(ga_ + j_ * 16) : "memory");   \
        }                                                                         \
    }

    // Q A-fragments with 1/sqrt(HS) folded in once (exact: *2^-3 in bf16)
    FragBF aq0, aq1;
    {
        const uint4* qr = (const uint4*)(qp + (size_t)(qbase + l16) * HS_);
        aq0.u[0] = qr[half*2];     aq0.u[1] = qr[half*2 + 1];
        aq1.u[0] = qr[4 + half*2]; aq1.u[1] = qr[5 + half*2];
        #pragma unroll
        for (int j = 0; j < 16; ++j) {
            aq0.v[j] = (__bf16)(SCALE_ * (float)aq0.v[j]);
            aq1.v[j] = (__bf16)(SCALE_ * (float)aq1.v[j]);
        }
    }

    float rmax[8], rsum[8];
    v8f acc0 = {}, acc1 = {}, acc2 = {}, acc3 = {};
    #pragma unroll
    for (int i = 0; i < 8; ++i) { rmax[i] = -INFINITY; rsum[i] = 0.f; }

    ISSUE_VTILE(0, 0)                       // prologue: tile 0 -> slot 0

    for (int kt = 0; kt < L_; kt += 32) {
        const int slot = (kt >> 5) & 1;
        // prefetch next V tile into other slot (wraps on last iter: keeps the
        // <=8-outstanding invariant so the wait below always drains this tile)
        ISSUE_VTILE(slot ^ 1, (kt + 32) & (L_ - 1))

        // K B-fragments: column = key index, K-dim = head dim (contiguous rows)
        FragBF b00, b01, b10, b11;
        {
            const uint4* kr0 = (const uint4*)(qp + (size_t)(kt + l16) * HS_);
            const uint4* kr1 = (const uint4*)(qp + (size_t)(kt + 16 + l16) * HS_);
            b00.u[0] = kr0[half*2];   b00.u[1] = kr0[half*2+1];
            b01.u[0] = kr0[4+half*2]; b01.u[1] = kr0[5+half*2];
            b10.u[0] = kr1[half*2];   b10.u[1] = kr1[half*2+1];
            b11.u[0] = kr1[4+half*2]; b11.u[1] = kr1[5+half*2];
        }

        // S = (Q*scale) K^T  (two 16x16 tiles covering 32 keys)
        v8f s0 = {}, s1 = {};
        s0 = __builtin_amdgcn_wmma_f32_16x16x32_bf16(false, aq0.v, false, b00.v, (short)0, s0, false, false);
        s0 = __builtin_amdgcn_wmma_f32_16x16x32_bf16(false, aq1.v, false, b01.v, (short)0, s0, false, false);
        s1 = __builtin_amdgcn_wmma_f32_16x16x32_bf16(false, aq0.v, false, b10.v, (short)0, s1, false, false);
        s1 = __builtin_amdgcn_wmma_f32_16x16x32_bf16(false, aq1.v, false, b11.v, (short)0, s1, false, false);

        // online softmax; rows of a C tile live in 16-lane halves (DPP reductions)
        #pragma unroll
        for (int i = 0; i < 8; ++i) {
            const float e0 = s0[i], e1 = s1[i];
            const float t  = row16_max(fmaxf(e0, e1));
            const float mo = rmax[i];
            const float mn = fmaxf(mo, t);
            rmax[i] = mn;
            const float alpha = __expf(mo - mn);
            rsum[i] *= alpha;
            acc0[i] *= alpha; acc1[i] *= alpha; acc2[i] *= alpha; acc3[i] *= alpha;
            const float p0 = __expf(e0 - mn);
            const float p1 = __expf(e1 - mn);
            rsum[i] += row16_sum(p0 + p1);
            const int r = i + half * 8;
            sP[w][r][l16]      = (__bf16)p0;   // C-layout -> row-major P in LDS
            sP[w][r][16 + l16] = (__bf16)p1;
        }

        // current V tile resident (async completes in order; newest 8 may fly)
        asm volatile("s_wait_asynccnt 0x8" ::: "memory");

        // P A-fragment (16x32, row = l16, keys half*16 .. half*16+15)
        FragBF pa;
        {
            const uint4* pr = (const uint4*)&sP[w][l16][0];
            pa.u[0] = pr[half*2]; pa.u[1] = pr[half*2 + 1];
        }
        // V B-fragments via LDS transpose: B[k][n] = V[k][n], k = half*16 + j
        v16bf vb0, vb1, vb2, vb3;
        #pragma unroll
        for (int j = 0; j < 16; ++j) {
            const int k = half * 16 + j;
            vb0[j] = sV[w][slot][k][ 0 + l16];
            vb1[j] = sV[w][slot][k][16 + l16];
            vb2[j] = sV[w][slot][k][32 + l16];
            vb3[j] = sV[w][slot][k][48 + l16];
        }
        acc0 = __builtin_amdgcn_wmma_f32_16x16x32_bf16(false, pa.v, false, vb0, (short)0, acc0, false, false);
        acc1 = __builtin_amdgcn_wmma_f32_16x16x32_bf16(false, pa.v, false, vb1, (short)0, acc1, false, false);
        acc2 = __builtin_amdgcn_wmma_f32_16x16x32_bf16(false, pa.v, false, vb2, (short)0, acc2, false, false);
        acc3 = __builtin_amdgcn_wmma_f32_16x16x32_bf16(false, pa.v, false, vb3, (short)0, acc3, false, false);
    }
#undef ISSUE_VTILE

    // epilogue: normalize and write bf16 attn output in [B, L, D] layout
    #pragma unroll
    for (int i = 0; i < 8; ++i) {
        const float inv = 1.f / rsum[i];
        const int r = qbase + i + half * 8;
        __bf16* orow = O + ((size_t)b * L_ + r) * D_ + h * HS_;
        orow[ 0 + l16] = (__bf16)(acc0[i] * inv);
        orow[16 + l16] = (__bf16)(acc1[i] * inv);
        orow[32 + l16] = (__bf16)(acc2[i] * inv);
        orow[48 + l16] = (__bf16)(acc3[i] * inv);
    }
}

// ---------------------------------------------------------------------------
extern "C" void kernel_launch(void* const* d_in, const int* in_sizes, int n_in,
                              void* d_out, int out_size, void* d_ws, size_t ws_size,
                              hipStream_t stream) {
    (void)in_sizes; (void)n_in; (void)out_size; (void)ws_size;
    const float* x  = (const float*)d_in[0];
    const float* Wq = (const float*)d_in[1];
    const float* Wo = (const float*)d_in[2];
    float* out = (float*)d_out;

    const size_t NX = (size_t)B_ * L_ * D_;   // 6,291,456
    const size_t NW = (size_t)D_ * D_;        //   589,824

    __bf16* qh    = (__bf16*)d_ws;            // [B,H,L,HS] bf16
    __bf16* xh    = qh + NX;                  // x in bf16 (dead after qproj)
    __bf16* attnh = xh;                       // aliases xh: same size, live later
    __bf16* wqh   = xh + NX;
    __bf16* woh   = wqh + NW;                 // total ws: ~27.5 MB

    cvt_bf16_kernel<<<(int)(NX / 8 / 256), 256, 0, stream>>>(x,  xh,  (int)(NX / 8));
    cvt_bf16_kernel<<<(int)(NW / 8 / 256), 256, 0, stream>>>(Wq, wqh, (int)(NW / 8));
    cvt_bf16_kernel<<<(int)(NW / 8 / 256), 256, 0, stream>>>(Wo, woh, (int)(NW / 8));

    const int gemm_blocks = ((B_ * L_) / 16) * (D_ / 64) / 8;   // 768
    gemm16x64_kernel<true><<<gemm_blocks, 256, 0, stream>>>(xh, wqh, qh, nullptr);

    const int fa_blocks = (B_ * H_ * (L_ / 16)) / FA_WAVES;     // 1536
    flash_kernel<<<fa_blocks, 32 * FA_WAVES, 0, stream>>>(qh, attnh);

    gemm16x64_kernel<false><<<gemm_blocks, 256, 0, stream>>>(attnh, woh, nullptr, out);
}
